// SymbolicDecoder_81054622810355
// MI455X (gfx1250) — matrix operand order repaired and checked by reading, not compiled, not run on hardware
//
#include <hip/hip_runtime.h>

typedef __attribute__((ext_vector_type(16))) __bf16 v16bf;
typedef __attribute__((ext_vector_type(8)))  __bf16 v8bf;
typedef __attribute__((ext_vector_type(4)))  __bf16 v4bf;
typedef __attribute__((ext_vector_type(8)))  float  v8f;

#define D 1024
#define BS 4096
#define MT 64          // batch rows per workgroup
#define KS 256         // K elements staged per round (64 rows x 256 K bf16 = 32 KB)

// task tables
__device__ __constant__ int c_snet[8] = {0,0,0,1,2,3,3,3};
__device__ __constant__ int c_sidx[8] = {0,1,2,6,6,0,1,2};
__device__ __constant__ int c_scol[8] = {0,1,2,3,4,19,20,21};
__device__ __constant__ int c_p1[14]  = {0,0,0,0,1,1,1,1,2,2,2,2,4,4};
__device__ __constant__ int c_p2[14]  = {3,4,5,6,3,4,5,6,3,4,5,6,5,6};

// ---- prep: fp32 feat -> bf16 (contiguous) ----
__global__ __launch_bounds__(256) void cvt_feat_kernel(const float* __restrict__ src,
                                                       __bf16* __restrict__ dst) {
  size_t i = ((size_t)blockIdx.x * 256 + threadIdx.x) * 4;
  float4 v = *(const float4*)(src + i);
  v4bf o = { (__bf16)v.x, (__bf16)v.y, (__bf16)v.z, (__bf16)v.w };
  *(v4bf*)(dst + i) = o;
}

// ---- prep: fp32 (K,N) -> bf16 (N,K) transpose ----
__global__ __launch_bounds__(256) void transpose_bf16_kernel(const float* __restrict__ src,
                                                             __bf16* __restrict__ dst,
                                                             int K, int N) {
  __shared__ float tile[32][33];
  int tx = threadIdx.x, ty = threadIdx.y;
  int n0 = blockIdx.x * 32, k0 = blockIdx.y * 32;
  #pragma unroll
  for (int i = 0; i < 32; i += 8)
    tile[ty + i][tx] = src[(size_t)(k0 + ty + i) * N + (n0 + tx)];
  __syncthreads();
  #pragma unroll
  for (int i = 0; i < 32; i += 8)
    dst[(size_t)(n0 + ty + i) * K + (k0 + tx)] = (__bf16)tile[tx][ty + i];
}

// ---- main fused MLP: layer1 (WMMA bf16) -> relu -> layer2 (1024->2) ----
__global__ __launch_bounds__(256) void fused_mlp_kernel(
    const __bf16* __restrict__ featb,   // (4096,7,1024) bf16
    const __bf16* __restrict__ sw1t,    // (4,1024N,1024K) bf16
    const __bf16* __restrict__ dw1t,    // (1024N,2048K) bf16
    const float* __restrict__ sb1, const float* __restrict__ sw2, const float* __restrict__ sb2,
    const float* __restrict__ db1, const float* __restrict__ dw2, const float* __restrict__ db2,
    float* __restrict__ out)            // (4096,22,2) fp32
{
  __shared__ __bf16 Alds[MT * KS];      // 32 KB A slab (64 rows x 256 K)
  __shared__ float out_lds[2 * MT];     // 64 rows x 2 outputs

  const int tid    = threadIdx.x;
  const int lane   = tid & 31;
  const int wave   = tid >> 5;
  const int lane16 = lane & 15;
  const int lhalf  = lane >> 4;
  const int mw     = wave >> 1;         // 4 M-subtiles of 16 rows
  const int nw     = wave & 1;          // 2 N-halves of 512 hidden cols
  const int m0     = blockIdx.x * MT;
  const int task   = blockIdx.y;

  const __bf16* w1t;
  const float *b1, *w2, *b2;
  int Ktot, ridx0, ridx1 = 0, col;
  if (task < 8) {
    int net = c_snet[task];
    w1t = sw1t + (size_t)net * D * D;
    b1 = sb1 + net * D;  w2 = sw2 + net * 2 * D;  b2 = sb2 + net * 2;
    Ktot = D; ridx0 = c_sidx[task]; ridx1 = c_sidx[task]; col = c_scol[task];
  } else {
    int p = task - 8;
    w1t = dw1t;  b1 = db1;  w2 = dw2;  b2 = db2;
    Ktot = 2 * D; ridx0 = c_p1[p]; ridx1 = c_p2[p]; col = 5 + p;
  }
  const int nstage = Ktot / KS;         // 4 (singles) or 8 (pairs)

  if (tid < 2 * MT) out_lds[tid] = 0.0f;

  float acc0[8] = {0,0,0,0,0,0,0,0};    // persistent layer-2 partials (2 outputs x 8 M-slots)
  float acc1[8] = {0,0,0,0,0,0,0,0};

  for (int cg = 0; cg < 4; ++cg) {      // 4 chunk-groups of 8 hidden 16-col chunks
    v8f c[8] = {};

    for (int s = 0; s < nstage; ++s) {
      const int kbase = s * KS;
      const int fidx  = (kbase >= D) ? ridx1 : ridx0;
      const int koff  = kbase & (D - 1);
      // cooperative stage of 64 rows x 256 K bf16 into LDS
      #pragma unroll
      for (int it = 0; it < 8; ++it) {
        int e   = (tid + it * 256) * 8;          // [0, 16384)
        int row = e >> 8;
        int k   = e & 255;
        const v8bf* src = (const v8bf*)(featb + ((size_t)(m0 + row) * 7 + fidx) * D + koff + k);
        *(v8bf*)(&Alds[e]) = *src;
      }
      __syncthreads();

      const __bf16* arow = Alds + (mw * 16 + lane16) * KS + lhalf * 16;
      #pragma unroll
      for (int i = 0; i < 8; ++i) {
        const int nbase = 16 * (nw * 32 + cg * 8 + i);
        const __bf16* brow = w1t + (size_t)(nbase + lane16) * Ktot + kbase + lhalf * 16;
        #pragma unroll 4
        for (int k = 0; k < KS; k += 32) {
          union { v16bf v; v8bf h[2]; } a;
          a.h[0] = *(const v8bf*)(arow + k);        // K +{0..7} / +{16..23}
          a.h[1] = *(const v8bf*)(arow + k + 8);    // K +{8..15} / +{24..31}
          v16bf b = *(const v16bf*)(brow + k);      // K-contiguous column of W1^T
          c[i] = __builtin_amdgcn_wmma_f32_16x16x32_bf16(
                     false, a.v, false, b, (short)0, c[i], false, false);
        }
      }
      __syncthreads();
    }

    // epilogue for this chunk-group: +b1, relu, contract against W2 (-> 2 cols)
    #pragma unroll
    for (int i = 0; i < 8; ++i) {
      int n = 16 * (nw * 32 + cg * 8 + i) + lane16;
      float b1n = b1[n];
      float w20 = w2[2 * n], w21 = w2[2 * n + 1];
      #pragma unroll
      for (int r = 0; r < 8; ++r) {
        float h = fmaxf(c[i][r] + b1n, 0.0f);
        acc0[r] += h * w20;
        acc1[r] += h * w21;
      }
    }
  }

  // cross-lane / cross-wave reduction in LDS
  #pragma unroll
  for (int r = 0; r < 8; ++r) {
    int m = mw * 16 + lhalf * 8 + r;    // C/D layout: VGPR r, lane half selects M+8
    atomicAdd(&out_lds[m * 2 + 0], acc0[r]);
    atomicAdd(&out_lds[m * 2 + 1], acc1[r]);
  }
  __syncthreads();

  if (tid < 2 * MT) {
    int j = tid & 1;
    out[(size_t)(m0 + (tid >> 1)) * 44 + col * 2 + j] = out_lds[tid] + b2[j];
  }
}

extern "C" void kernel_launch(void* const* d_in, const int* in_sizes, int n_in,
                              void* d_out, int out_size, void* d_ws, size_t ws_size,
                              hipStream_t stream) {
  (void)in_sizes; (void)n_in; (void)out_size;
  const float* feat = (const float*)d_in[0];
  const float* sw1  = (const float*)d_in[1];
  const float* sb1  = (const float*)d_in[2];
  const float* sw2  = (const float*)d_in[3];
  const float* sb2  = (const float*)d_in[4];
  const float* dw1  = (const float*)d_in[5];
  const float* db1  = (const float*)d_in[6];
  const float* dw2  = (const float*)d_in[7];
  const float* db2  = (const float*)d_in[8];
  float* out = (float*)d_out;

  char* ws = (char*)d_ws;
  const size_t featb_bytes = (size_t)BS * 7 * D * 2;   // 58,720,256
  const size_t sw1t_bytes  = (size_t)4 * D * D * 2;    //  8,388,608
  __bf16* featb = (__bf16*)ws;
  __bf16* sw1t  = (__bf16*)(ws + featb_bytes);
  __bf16* dw1t  = (__bf16*)(ws + featb_bytes + sw1t_bytes);
  if (ws_size < featb_bytes + sw1t_bytes + (size_t)D * 2 * D * 2) return;

  // feat -> bf16: 29,360,128 elems / 4 per thread / 256 per block
  cvt_feat_kernel<<<28672, 256, 0, stream>>>(feat, featb);
  // weight transposes (K,N) fp32 -> (N,K) bf16
  for (int net = 0; net < 4; ++net)
    transpose_bf16_kernel<<<dim3(32, 32), dim3(32, 8), 0, stream>>>(
        sw1 + (size_t)net * D * D, sw1t + (size_t)net * D * D, D, D);
  transpose_bf16_kernel<<<dim3(32, 64), dim3(32, 8), 0, stream>>>(dw1, dw1t, 2 * D, D);

  // fused MLP: 64 row-blocks x 22 heads
  fused_mlp_kernel<<<dim3(64, 22), 256, 0, stream>>>(
      featb, sw1t, dw1t, sb1, sw2, sb2, db1, dw2, db2, out);
}
